// SRNN_49125835931746
// MI455X (gfx1250) — matrix operand order repaired and checked by reading, not compiled, not run on hardware
//
#include <hip/hip_runtime.h>
#include <cstddef>

typedef __attribute__((ext_vector_type(16))) _Float16 v16h;
typedef __attribute__((ext_vector_type(8)))  _Float16 v8h;
typedef __attribute__((ext_vector_type(8)))  float    v8f;
typedef __attribute__((ext_vector_type(4)))  float    f32x4;

#define N_T   250
#define N_Bb  256
#define N_IN  700
#define N_OUTo 20
#define NR1   120
#define NR2   84
#define NR1P  128
#define NR2P  96
#define KT_FULL 21      // k-tiles 0..20 fully inside K=700
#define KT_ALL  22
#define ALPHA_F 0.9512294245007141f
#define KAPPA_F 0.9512294245007141f
#define THR_F   0.6f
#define MT_PER_BLK 8

__device__ __forceinline__ v8f zero8() {
    v8f z;
#pragma unroll
    for (int i = 0; i < 8; ++i) z[i] = 0.0f;
    return z;
}

// ---------------------------------------------------------------------------
// Kernel 1: xproj = x @ w1^T, stored directly in WMMA C-fragment layout:
//   xproj[((mtile*8 + wave)*32 + lane)*8 + r]   (mtile = m/16, m = t*256+b)
// B-fragments (w1) hoisted into registers above the m-tile loop.
// ---------------------------------------------------------------------------
__global__ __launch_bounds__(256)
void xproj_kernel(const float* __restrict__ x, const float* __restrict__ w1,
                  float* __restrict__ xproj) {
    const int lane = threadIdx.x & 31;
    const int wave = threadIdx.x >> 5;          // 0..7 -> N tile
    const int n0   = wave * 16;

    const int am   = lane & 15;                 // A row within tile
    const int a_kb = (lane >> 4) ? 8 : 0;       // A K base per half-wave
    const int bn   = lane & 15;                 // B column within tile
    const int b_kb = (lane >> 4) ? 8 : 0;       // B K base per half-wave

    const int   j     = n0 + bn;                // w1 row (output neuron)
    const int   jc    = (j < NR1) ? j : (NR1 - 1);
    const float jmask = (j < NR1) ? 1.0f : 0.0f;
    const float* wrow = w1 + (size_t)jc * N_IN;

    // ---- one-time build of all 22 B-fragments (m-invariant) ----
    v16h bfrag[KT_ALL];
#pragma unroll
    for (int kt = 0; kt < KT_FULL; ++kt) {
        const int k0 = kt * 32;
        const f32x4 b0 = *(const f32x4*)(wrow + k0 + b_kb);
        const f32x4 b1 = *(const f32x4*)(wrow + k0 + b_kb + 4);
        const f32x4 b2 = *(const f32x4*)(wrow + k0 + b_kb + 16);
        const f32x4 b3 = *(const f32x4*)(wrow + k0 + b_kb + 20);
        v16h b;
#pragma unroll
        for (int h = 0; h < 4; ++h) {
            b[h]      = (_Float16)(b0[h] * jmask);
            b[4 + h]  = (_Float16)(b1[h] * jmask);
            b[8 + h]  = (_Float16)(b2[h] * jmask);
            b[12 + h] = (_Float16)(b3[h] * jmask);
        }
        bfrag[kt] = b;
    }
    {   // tail B tile (k 672..703): clamp index, select value
        const int k0 = KT_FULL * 32;
        v16h b;
#pragma unroll
        for (int h = 0; h < 16; ++h) {
            const int k  = k0 + b_kb + h + (h >= 8 ? 8 : 0);
            const int kc = (k < N_IN) ? k : (N_IN - 1);
            const float v = wrow[kc] * jmask;
            b[h] = (_Float16)((k < N_IN) ? v : 0.0f);
        }
        bfrag[KT_FULL] = b;
    }

    // ---- m-tile loop: A loads + wmma only ----
    for (int mt = 0; mt < MT_PER_BLK; ++mt) {
        const size_t mtile = (size_t)blockIdx.x * MT_PER_BLK + mt;
        const int m0 = (int)mtile * 16;
        const float* xrow = x + (size_t)(m0 + am) * N_IN;

        v8f acc = zero8();
#pragma unroll
        for (int kt = 0; kt < KT_FULL; ++kt) {
            const int k0 = kt * 32;
            const f32x4 a0 = *(const f32x4*)(xrow + k0 + a_kb);
            const f32x4 a1 = *(const f32x4*)(xrow + k0 + a_kb + 4);
            const f32x4 a2 = *(const f32x4*)(xrow + k0 + a_kb + 16);
            const f32x4 a3 = *(const f32x4*)(xrow + k0 + a_kb + 20);
            v16h a;
#pragma unroll
            for (int h = 0; h < 4; ++h) {
                a[h]      = (_Float16)a0[h];
                a[4 + h]  = (_Float16)a1[h];
                a[8 + h]  = (_Float16)a2[h];
                a[12 + h] = (_Float16)a3[h];
            }
            acc = __builtin_amdgcn_wmma_f32_16x16x32_f16(
                false, a, false, bfrag[kt], (short)0, acc, false, false);
        }
        {   // tail A tile
            const int k0 = KT_FULL * 32;
            v16h a;
#pragma unroll
            for (int h = 0; h < 8; ++h) {
                const int k1 = k0 + a_kb + h;          // <= 687, always valid
                const int k2 = k1 + 16;                // may reach 703
                const int k2c = (k2 < N_IN) ? k2 : (N_IN - 1);
                const float v2 = xrow[k2c];
                a[h]     = (_Float16)xrow[k1];
                a[h + 8] = (_Float16)((k2 < N_IN) ? v2 : 0.0f);
            }
            acc = __builtin_amdgcn_wmma_f32_16x16x32_f16(
                false, a, false, bfrag[KT_FULL], (short)0, acc, false, false);
        }

        // fragment-layout store: 32 contiguous bytes per lane
        float* op = xproj + (((mtile * 8 + wave) * 32) + lane) * 8;
        f32x4 lo, hi;
#pragma unroll
        for (int h = 0; h < 4; ++h) { lo[h] = acc[h]; hi[h] = acc[4 + h]; }
        *(f32x4*)op       = lo;
        *(f32x4*)(op + 4) = hi;
    }
}

// ---------------------------------------------------------------------------
// Kernel 2: persistent per-batch-tile scan. 16 blocks x 256 threads (8 waves).
// ---------------------------------------------------------------------------
__global__ __launch_bounds__(256)
void srnn_scan_kernel(const float* __restrict__ xproj,
                      const float* __restrict__ wrec1,
                      const float* __restrict__ w2,
                      const float* __restrict__ wrec2,
                      const float* __restrict__ wout,
                      float* __restrict__ out) {
    __shared__ alignas(16) _Float16 z1s[2][16][NR1P];   // 8 KB, row = 256 B
    __shared__ alignas(16) _Float16 z2s[2][16][NR2P];   // 6 KB, row = 192 B
    __shared__ alignas(16) float    vos[16][32];        // 2 KB

    const int lane = threadIdx.x & 31;
    const int wave = threadIdx.x >> 5;
    const int n0   = wave * 16;
    const int b0   = blockIdx.x * 16;

    const int am   = lane & 15;
    const int a_kb = (lane >> 4) ? 8 : 0;
    const int b_kb = (lane >> 4) ? 8 : 0;
    const int cn   = lane & 15;
    const int mb   = (lane >> 4) ? 8 : 0;
    const int j    = n0 + cn;

    // ---- preload weight B-fragments into registers (fixed per wave) ----
    auto loadB = [&](const float* W, int ldw, int nrows, int k0,
                     bool zero_diag) -> v16h {
        v16h b;
#pragma unroll
        for (int h = 0; h < 16; ++h) {
            const int k = k0 + b_kb + h + (h >= 8 ? 8 : 0);
            float v = 0.0f;
            if (j < nrows && k < ldw) v = W[(size_t)j * ldw + k];
            if (zero_diag && j == k) v = 0.0f;   // forward() zeroes diagonal
            b[h] = (_Float16)v;
        }
        return b;
    };

    v16h wb1[4], wb2[4], wb3[3], wb4[3];
#pragma unroll
    for (int kt = 0; kt < 4; ++kt) wb1[kt] = loadB(wrec1, NR1, NR1, kt * 32, true);
#pragma unroll
    for (int kt = 0; kt < 4; ++kt) wb2[kt] = loadB(w2, NR1, NR2, kt * 32, false);
#pragma unroll
    for (int kt = 0; kt < 3; ++kt) wb3[kt] = loadB(wrec2, NR2, NR2, kt * 32, true);
#pragma unroll
    for (int kt = 0; kt < 3; ++kt) wb4[kt] = loadB(wout, NR2, N_OUTo, kt * 32, false);

    // ---- zero spike buffers; out[0] = softmax(0) = 0.05 ----
    {
        unsigned int* z1w = (unsigned int*)z1s;
        unsigned int* z2w = (unsigned int*)z2s;
        for (int i = threadIdx.x; i < 2 * 16 * NR1P / 2; i += 256) z1w[i] = 0u;
        for (int i = threadIdx.x; i < 2 * 16 * NR2P / 2; i += 256) z2w[i] = 0u;
    }
    if (threadIdx.x < 16) {
        float* op = out + (size_t)(b0 + threadIdx.x) * N_OUTo;
#pragma unroll
        for (int o = 0; o < N_OUTo; ++o) op[o] = 0.05f;
    }
    __syncthreads();

    auto loadA1 = [&](int buf, int k0) -> v16h {
        const v8h lo = *(const v8h*)&z1s[buf][am][k0 + a_kb];
        const v8h hi = *(const v8h*)&z1s[buf][am][k0 + a_kb + 16];
        v16h a;
#pragma unroll
        for (int h = 0; h < 8; ++h) { a[h] = lo[h]; a[h + 8] = hi[h]; }
        return a;
    };
    auto loadA2 = [&](int buf, int k0) -> v16h {
        const v8h lo = *(const v8h*)&z2s[buf][am][k0 + a_kb];
        const v8h hi = *(const v8h*)&z2s[buf][am][k0 + a_kb + 16];
        v16h a;
#pragma unroll
        for (int h = 0; h < 8; ++h) { a[h] = lo[h]; a[h + 8] = hi[h]; }
        return a;
    };

    v8f v1a = zero8(), v2a = zero8(), voa = zero8();

    for (int t = 0; t < N_T - 1; ++t) {
        const int p = t & 1, q = p ^ 1;
        // fragment-layout xproj block for (t, this batch tile, this wave)
        const float* xpf = xproj +
            ((((size_t)(t * 16 + blockIdx.x)) * 8 + wave) * 32 + lane) * 8;

        // ---- phase 1 (all 8 waves): v1 update + spikes ----
        const f32x4 x0 = *(const f32x4*)xpf;
        const f32x4 x1 = *(const f32x4*)(xpf + 4);
        v8f acc;
#pragma unroll
        for (int r = 0; r < 8; ++r) {
            const float z1old = (float)z1s[p][mb + r][n0 + cn];
            const float xv = (r < 4) ? x0[r] : x1[r - 4];
            acc[r] = ALPHA_F * v1a[r] + xv - THR_F * z1old;
        }
#pragma unroll
        for (int kt = 0; kt < 4; ++kt)
            acc = __builtin_amdgcn_wmma_f32_16x16x32_f16(
                false, loadA1(p, kt * 32), false, wb1[kt], (short)0, acc, false, false);
        v1a = acc;
#pragma unroll
        for (int r = 0; r < 8; ++r)
            z1s[q][mb + r][n0 + cn] = (_Float16)((acc[r] > THR_F) ? 1.0f : 0.0f);
        __builtin_prefetch(xpf + (size_t)16 * 8 * 32 * 8, 0, 0);  // next t
        __syncthreads();

        // ---- phase 2 (waves 0..5): v2 update + spikes ----
        if (wave < 6) {
            v8f a2;
#pragma unroll
            for (int r = 0; r < 8; ++r) {
                const float z2old = (float)z2s[p][mb + r][n0 + cn];
                a2[r] = ALPHA_F * v2a[r] - THR_F * z2old;
            }
#pragma unroll
            for (int kt = 0; kt < 4; ++kt)   // z1_new @ w2^T   (K=128)
                a2 = __builtin_amdgcn_wmma_f32_16x16x32_f16(
                    false, loadA1(q, kt * 32), false, wb2[kt], (short)0, a2, false, false);
#pragma unroll
            for (int kt = 0; kt < 3; ++kt)   // z2_old @ wrec2^T (K=96)
                a2 = __builtin_amdgcn_wmma_f32_16x16x32_f16(
                    false, loadA2(p, kt * 32), false, wb3[kt], (short)0, a2, false, false);
            v2a = a2;
#pragma unroll
            for (int r = 0; r < 8; ++r)
                z2s[q][mb + r][n0 + cn] = (_Float16)((a2[r] > THR_F) ? 1.0f : 0.0f);
        }
        __syncthreads();

        // ---- phase 3 (waves 0..1): leaky readout ----
        if (wave < 2) {
            v8f vo;
#pragma unroll
            for (int r = 0; r < 8; ++r) vo[r] = KAPPA_F * voa[r];
#pragma unroll
            for (int kt = 0; kt < 3; ++kt)   // z2_new @ wout^T (K=96)
                vo = __builtin_amdgcn_wmma_f32_16x16x32_f16(
                    false, loadA2(q, kt * 32), false, wb4[kt], (short)0, vo, false, false);
            voa = vo;
#pragma unroll
            for (int r = 0; r < 8; ++r) vos[mb + r][n0 + cn] = vo[r];
        }
        __syncthreads();

        // ---- phase 4: per-row softmax over 20 outputs ----
        if (threadIdx.x < 16) {
            const int m = threadIdx.x;
            float mx = vos[m][0];
#pragma unroll
            for (int o = 1; o < N_OUTo; ++o) mx = fmaxf(mx, vos[m][o]);
            float e[N_OUTo];
            float s = 0.0f;
#pragma unroll
            for (int o = 0; o < N_OUTo; ++o) { e[o] = __expf(vos[m][o] - mx); s += e[o]; }
            const float inv = 1.0f / s;
            float* op = out + ((size_t)(t + 1) * N_Bb + b0 + m) * N_OUTo;
#pragma unroll
            for (int o = 0; o < N_OUTo; ++o) op[o] = e[o] * inv;
        }
    }
}

// ---------------------------------------------------------------------------
extern "C" void kernel_launch(void* const* d_in, const int* in_sizes, int n_in,
                              void* d_out, int out_size, void* d_ws, size_t ws_size,
                              hipStream_t stream) {
    (void)in_sizes; (void)n_in; (void)out_size; (void)ws_size;
    const float* x     = (const float*)d_in[0];
    // d_in[1] = yt (unused), d_in[7] = do_training (unused)
    const float* w1    = (const float*)d_in[2];
    const float* wrec1 = (const float*)d_in[3];
    const float* w2    = (const float*)d_in[4];
    const float* wrec2 = (const float*)d_in[5];
    const float* wout  = (const float*)d_in[6];
    float* out   = (float*)d_out;
    float* xproj = (float*)d_ws;   // 4000 mtiles * 8 * 32 * 8 f32 = 32.768 MB

    dim3 blk(256);
    xproj_kernel<<<dim3(500), blk, 0, stream>>>(x, w1, xproj);
    srnn_scan_kernel<<<dim3(16), blk, 0, stream>>>(xproj, wrec1, w2, wrec2, wout, out);
}
